// MHFSpectralConv_12841952215734
// MI455X (gfx1250) — compile-verified, matrix operand order checked by reading, not run on hardware
//
#include <hip/hip_runtime.h>

// ---------------------------------------------------------------------------
// MHF spectral conv for MI455X (gfx1250, wave32, WMMA).
// Entire pipeline expressed as bf16 GEMMs through one WMMA kernel:
//   x(f32) -> (DFT_W gemm, fused f32->bf16) -> (DFT_H gemm) -> combine
//     -> (mode-mix gemm, 1024 batch) -> pack -> (iDFT_H gemm)
//     -> (iDFT_W gemm) -> (fc gemm + bias, f32 out)
// ---------------------------------------------------------------------------

typedef __attribute__((ext_vector_type(16))) __bf16 v16bf;
typedef __attribute__((ext_vector_type(8)))  __bf16 v8bf;
typedef __attribute__((ext_vector_type(8)))  float  v8f;
typedef __attribute__((ext_vector_type(4)))  float  v4f;

#define TWO_PI 6.28318530717958647692f

// ---------------------------------------------------------------------------
// Generic batched GEMM:  C[z] = A[z][M x K] * B[z][K x N]  (row-major)
// BM=BN=64, BK=32, 256 threads = 8 wave32 waves, each wave -> 16x32 output.
// A side: bf16 (staged via global_load_async_to_lds_b128, ASYNCcnt-tracked)
//         or f32 (converted to bf16 during staging -> saves a full pre-pass).
// B tile stored transposed in LDS ([n][k]) so both fragment loads are
// contiguous 16B ds_loads matching the 16-bit WMMA operand layout:
//   lane<16  : K in {0..7, 16..23},  lane>=16 : K in {8..15, 24..31}.
// Requirements used here: K % 32 == 0, N % 64 == 0 (M may be ragged; ragged
// rows are garbage in LDS for the async path but only feed epilogue-masked
// output rows, since C[m,:] depends only on A[m,:]).
// ---------------------------------------------------------------------------
template <bool AF32, bool OUTF32, bool BIAS>
__global__ __launch_bounds__(256) void gemm_bf16_wmma(
    const void* __restrict__ A, long long aStride, int lda,
    const __bf16* __restrict__ B, long long bStride, int ldb,
    void* __restrict__ Cv, long long cStride, int ldc,
    const float* __restrict__ bias, int M, int N, int K) {
  __shared__ __bf16 As[64][40];  // +8 pad: rows stay 16B aligned, fewer conflicts
  __shared__ __bf16 Bs[64][40];  // transposed: Bs[n][k]

  const int tid  = threadIdx.x;
  const int lane = tid & 31;
  const int wid  = tid >> 5;
  const int m0 = blockIdx.y * 64;
  const int n0 = blockIdx.x * 64;

  const __bf16* Abh = AF32 ? nullptr
                           : ((const __bf16*)A + (size_t)blockIdx.z * (size_t)aStride);
  const float*  Abf = AF32 ? ((const float*)A + (size_t)blockIdx.z * (size_t)aStride)
                           : nullptr;
  const __bf16* Bb = B + (size_t)blockIdx.z * (size_t)bStride;

  const int mbase = (wid & 3) * 16;   // 4 M-tiles
  const int nbase = (wid >> 2) * 32;  // 2 waves x 32 cols
  const int r16   = lane & 15;
  const int khalf = lane >> 4;

  v8f acc0 = {};
  v8f acc1 = {};

  const int arow = tid >> 2;       // 0..63
  const int achk = (tid & 3) * 8;  // 0,8,16,24 (k)
  const int brow = tid >> 3;       // 0..31 (k)
  const int bchk = (tid & 7) * 8;  // 0..56 (n)
  const bool arow_ok = (m0 + arow) < M;

  // LDS byte offset of this thread's A-tile chunk (flat LDS aperture: low 32
  // bits of a generic shared pointer are the LDS offset).
  const unsigned ldsA = (unsigned)(uintptr_t)(&As[arow][achk]);

  for (int k0 = 0; k0 < K; k0 += 32) {
    // --- stage A tile (64 x 32 bf16) ----------------------------------------
    if constexpr (AF32) {
      v4f lo = {}, hi = {};
      if (arow_ok) {
        const float* ap = Abf + (size_t)(m0 + arow) * lda + (k0 + achk);
        lo = *(const v4f*)ap;
        hi = *(const v4f*)(ap + 4);
      }
      v8bf av;
#pragma unroll
      for (int j = 0; j < 4; ++j) {
        av[j]     = (__bf16)lo[j];
        av[4 + j] = (__bf16)hi[j];
      }
      *(v8bf*)(&As[arow][achk]) = av;
      if (arow_ok && k0 + 32 < K)
        __builtin_prefetch(Abf + (size_t)(m0 + arow) * lda + (k0 + 32 + achk), 0, 3);
    } else {
      // CDNA5 async copy: global -> LDS without touching VGPRs (ASYNCcnt).
      if (arow_ok) {
        const unsigned long long gp =
            (unsigned long long)(Abh + (size_t)(m0 + arow) * lda + (k0 + achk));
        asm volatile("global_load_async_to_lds_b128 %0, %1, off"
                     :: "v"(ldsA), "v"(gp) : "memory");
      }
    }

    // --- stage B tile (32 x 64) transposed into LDS -------------------------
    v8bf bv = *(const v8bf*)(Bb + (size_t)(k0 + brow) * ldb + (n0 + bchk));
#pragma unroll
    for (int j = 0; j < 8; ++j) Bs[bchk + j][brow] = bv[j];
    if (k0 + 32 < K)
      __builtin_prefetch(Bb + (size_t)(k0 + 32 + brow) * ldb + (n0 + bchk), 0, 3);

    if constexpr (!AF32)
      asm volatile("s_wait_asynccnt 0x0" ::: "memory");
    __syncthreads();

    // --- fragments + WMMA ---------------------------------------------------
    union Frag { v8bf h[2]; v16bf v; };
    Frag fa, fb0, fb1;
    fa.h[0]  = *(const v8bf*)(&As[mbase + r16][khalf * 8]);
    fa.h[1]  = *(const v8bf*)(&As[mbase + r16][16 + khalf * 8]);
    fb0.h[0] = *(const v8bf*)(&Bs[nbase + r16][khalf * 8]);
    fb0.h[1] = *(const v8bf*)(&Bs[nbase + r16][16 + khalf * 8]);
    fb1.h[0] = *(const v8bf*)(&Bs[nbase + 16 + r16][khalf * 8]);
    fb1.h[1] = *(const v8bf*)(&Bs[nbase + 16 + r16][16 + khalf * 8]);

    acc0 = __builtin_amdgcn_wmma_f32_16x16x32_bf16(false, fa.v, false, fb0.v,
                                                   (short)0, acc0, false, false);
    acc1 = __builtin_amdgcn_wmma_f32_16x16x32_bf16(false, fa.v, false, fb1.v,
                                                   (short)0, acc1, false, false);
    __syncthreads();
  }

  // --- epilogue: C layout VGPR r -> M = r + 8*khalf, N = lane&15 ------------
  const int gm0 = m0 + mbase + khalf * 8;
  const int gn0 = n0 + nbase + r16;
  if (OUTF32) {
    float* C = (float*)Cv + (size_t)blockIdx.z * (size_t)cStride;
#pragma unroll
    for (int r = 0; r < 8; ++r) {
      const int gm = gm0 + r;
      if (gm < M) {
        const float bsv = BIAS ? bias[gm] : 0.0f;
        C[(size_t)gm * ldc + gn0]      = acc0[r] + bsv;
        C[(size_t)gm * ldc + gn0 + 16] = acc1[r] + bsv;
      }
    }
  } else {
    __bf16* C = (__bf16*)Cv + (size_t)blockIdx.z * (size_t)cStride;
#pragma unroll
    for (int r = 0; r < 8; ++r) {
      const int gm = gm0 + r;
      if (gm < M) {
        C[(size_t)gm * ldc + gn0]      = (__bf16)acc0[r];
        C[(size_t)gm * ldc + gn0 + 16] = (__bf16)acc1[r];
      }
    }
  }
}

// ---------------------------------------------------------------------------
// DFT coefficient tables (bf16), all normalization folded into Gw.
//   Fw  [256w x 64]: n<32: cos(2pi n w/256), n+32: -sin   (forward W: re|im)
//   Fh2 [64  x 256h]: m<32: cos(2pi m h/256), m+32: +sin  (forward H: C|S)
//   Ahs [256h x 64]: m<32: cos, m+32: sin                 (inverse H)
//   Gw  [64  x 256w]: n<32: a_n cos, n+32: -a_n sin, a_0=1/65536, a_n=2/65536
// ---------------------------------------------------------------------------
__global__ void k_tables(__bf16* __restrict__ Fw, __bf16* __restrict__ Fh2,
                         __bf16* __restrict__ Ahs, __bf16* __restrict__ Gw) {
  const int idx = blockIdx.x * 256 + threadIdx.x;  // 65536 total
  const int seg = idx >> 14;
  const int t   = idx & 16383;
  if (seg == 0) {
    const int wv = t >> 6, col = t & 63, n = col & 31;
    const float ang = TWO_PI * (float)(n * wv) * (1.0f / 256.0f);
    Fw[t] = (__bf16)((col < 32) ? cosf(ang) : -sinf(ang));
  } else if (seg == 1) {
    const int row = t >> 8, h = t & 255, m = row & 31;
    const float ang = TWO_PI * (float)(m * h) * (1.0f / 256.0f);
    Fh2[t] = (__bf16)((row < 32) ? cosf(ang) : sinf(ang));
  } else if (seg == 2) {
    const int h = t >> 6, col = t & 63, m = col & 31;
    const float ang = TWO_PI * (float)(m * h) * (1.0f / 256.0f);
    Ahs[t] = (__bf16)((col < 32) ? cosf(ang) : sinf(ang));
  } else {
    const int row = t >> 8, wv = t & 255, n = row & 31;
    const float a = ((n == 0) ? 1.0f : 2.0f) * (1.0f / 65536.0f);
    const float ang = TWO_PI * (float)(n * wv) * (1.0f / 256.0f);
    Gw[t] = (__bf16)((row < 32) ? a * cosf(ang) : -a * sinf(ang));
  }
}

__global__ void k_cvt_f32_bf16(const float* __restrict__ in,
                               __bf16* __restrict__ out, long long n) {
  const long long i = (long long)blockIdx.x * 256 + threadIdx.x;
  if (i < n) out[i] = (__bf16)in[i];
}

// T[img][64x64] (C@yr | C@yi ; S@yr | S@yi) -> per-mode A matrices
// Am[mode][16x128]: rows 0..7 = Re over batch, rows 8..15 = Im over batch.
__global__ void k_combine(const __bf16* __restrict__ T, __bf16* __restrict__ Am) {
  const int gid  = blockIdx.x * 256 + threadIdx.x;  // 1,048,576
  const int ci   = gid & 127;
  const int b    = (gid >> 7) & 7;
  const int mode = gid >> 10;
  const int m = mode >> 5, n = mode & 31;
  const __bf16* Ti = T + (size_t)(b * 128 + ci) * 4096;
  const float c_r = (float)Ti[m * 64 + n];
  const float c_i = (float)Ti[m * 64 + 32 + n];
  const float s_r = (float)Ti[(32 + m) * 64 + n];
  const float s_i = (float)Ti[(32 + m) * 64 + 32 + n];
  __bf16* dst = Am + (size_t)mode * 2048;
  dst[b * 128 + ci]       = (__bf16)(c_r + s_i);  // Re = C@yr + S@yi
  dst[(8 + b) * 128 + ci] = (__bf16)(c_i - s_r);  // Im = C@yi - S@yr
}

// weight[i][o][m][n] (f32) -> Wb[mode][i][o] (bf16), coalesced reads.
__global__ void k_weight_pack(const float* __restrict__ w, __bf16* __restrict__ Wb) {
  const long long gid = (long long)blockIdx.x * 256 + threadIdx.x;  // 16,777,216
  const int n = (int)(gid & 31);
  const int m = (int)((gid >> 5) & 31);
  const int o = (int)((gid >> 10) & 127);
  const int i = (int)(gid >> 17);
  Wb[(size_t)(m * 32 + n) * 16384 + (size_t)i * 128 + o] = (__bf16)w[gid];
}

// blk[mode][16x128] -> per-(b,o) B3[64x64] = [P ; -Q | Q ; P] stacking:
//   cols 0..31 : [P ; -Q],  cols 32..63 : [Q ; P]   (k rows: m then 32+m)
__global__ void k_pack_b3(const __bf16* __restrict__ blk, __bf16* __restrict__ B3) {
  const int gid = blockIdx.x * 256 + threadIdx.x;  // 4,194,304
  const int j   = gid & 63;
  const int k   = (gid >> 6) & 63;
  const int img = gid >> 12;            // img = b*128 + o
  const int b = img >> 7, o = img & 127;
  const int m = k & 31, top = k >> 5;
  const int n = j & 31, right = j >> 5;
  const size_t mb = (size_t)(m * 32 + n) * 2048;
  const float P = (float)blk[mb + b * 128 + o];
  const float Q = (float)blk[mb + (8 + b) * 128 + o];
  const float v = right ? (top ? P : Q) : (top ? -Q : P);
  B3[(size_t)img * 4096 + k * 64 + j] = (__bf16)v;
}

// ---------------------------------------------------------------------------
extern "C" void kernel_launch(void* const* d_in, const int* in_sizes, int n_in,
                              void* d_out, int out_size, void* d_ws, size_t ws_size,
                              hipStream_t stream) {
  const float* x   = (const float*)d_in[0];  // [8,128,256,256]
  const float* w   = (const float*)d_in[1];  // [1,128,128,32,32]
  const float* fcw = (const float*)d_in[2];  // [128,128]
  const float* fcb = (const float*)d_in[3];  // [128]
  (void)in_sizes; (void)n_in; (void)out_size; (void)ws_size;

  char* ws = (char*)d_ws;
  size_t off = 0;
  auto alloc = [&](size_t bytes) -> char* {
    char* p = ws + off;
    off += (bytes + 255) & ~(size_t)255;
    return p;
  };

  __bf16* tabFw  = (__bf16*)alloc(256 * 64 * 2);
  __bf16* tabFh2 = (__bf16*)alloc(64 * 256 * 2);
  __bf16* tabAhs = (__bf16*)alloc(256 * 64 * 2);
  __bf16* tabGw  = (__bf16*)alloc(64 * 256 * 2);
  __bf16* fcwb   = (__bf16*)alloc(128 * 128 * 2);
  __bf16* Y      = (__bf16*)alloc((size_t)16777216 * 2);  // [1024][256][64]
  __bf16* T      = (__bf16*)alloc((size_t)4194304 * 2);   // [1024][64][64]
  __bf16* Amod   = (__bf16*)alloc((size_t)2097152 * 2);   // [1024][16][128]
  __bf16* Wb     = (__bf16*)alloc((size_t)16777216 * 2);  // [1024][128][128]
  __bf16* blkS   = (__bf16*)alloc((size_t)2097152 * 2);   // [1024][16][128]
  __bf16* B3     = (__bf16*)alloc((size_t)4194304 * 2);   // [1024][64][64]
  __bf16* UV     = (__bf16*)alloc((size_t)16777216 * 2);  // [1024][256][64]
  __bf16* xr     = (__bf16*)alloc((size_t)67108864 * 2);  // [1024][256][256]

  // prep: tables, fc weight cast, spectral weight transpose+cast
  k_tables<<<256, 256, 0, stream>>>(tabFw, tabFh2, tabAhs, tabGw);
  k_cvt_f32_bf16<<<64, 256, 0, stream>>>(fcw, fcwb, 16384LL);
  k_weight_pack<<<65536, 256, 0, stream>>>(w, Wb);

  // 1a: forward DFT along W, f32 A consumed directly (fused bf16 convert).
  //     Y[img] = x[img] @ Fw   (M=256,K=256,N=64)
  gemm_bf16_wmma<true, false, false><<<dim3(1, 4, 1024), 256, 0, stream>>>(
      x, 65536LL, 256, tabFw, 0LL, 64, Y, 16384LL, 64, nullptr, 256, 64, 256);
  // 1b: forward DFT along H.  T[img] = [C;S] @ Y[img]  (M=64,K=256,N=64)
  gemm_bf16_wmma<false, false, false><<<dim3(1, 1, 1024), 256, 0, stream>>>(
      tabFh2, 0LL, 256, Y, 16384LL, 64, T, 4096LL, 64, nullptr, 64, 64, 256);
  k_combine<<<4096, 256, 0, stream>>>(T, Amod);

  // 2: per-mode channel mix. blk[mode] = Amod[mode] @ Wb[mode] (M=16,K=128,N=128)
  gemm_bf16_wmma<false, false, false><<<dim3(2, 1, 1024), 256, 0, stream>>>(
      Amod, 2048LL, 128, Wb, 16384LL, 128, blkS, 2048LL, 128, nullptr, 16, 128, 128);
  k_pack_b3<<<16384, 256, 0, stream>>>(blkS, B3);

  // 3a: inverse along H.  UV[img] = Ahs @ B3[img]   (M=256,K=64,N=64)
  gemm_bf16_wmma<false, false, false><<<dim3(1, 4, 1024), 256, 0, stream>>>(
      tabAhs, 0LL, 64, B3, 4096LL, 64, UV, 16384LL, 64, nullptr, 256, 64, 64);
  // 3b: inverse along W.  xr[img] = UV[img] @ Gw    (M=256,K=64,N=256)
  gemm_bf16_wmma<false, false, false><<<dim3(4, 4, 1024), 256, 0, stream>>>(
      UV, 16384LL, 64, tabGw, 0LL, 256, xr, 65536LL, 256, nullptr, 256, 256, 64);

  // 4: fc (1x1 conv) + bias, f32 out. out[b] = fcw @ xr[b] (M=128,K=128,N=65536)
  gemm_bf16_wmma<false, true, true><<<dim3(1024, 2, 8), 256, 0, stream>>>(
      fcwb, 0LL, 128, xr, 8388608LL, 65536, d_out, 8388608LL, 65536, fcb,
      128, 65536, 128);
}
